// RNNModel_50740743635348
// MI455X (gfx1250) — matrix operand order repaired and checked by reading, not compiled, not run on hardware
//
#include <hip/hip_runtime.h>

typedef __attribute__((ext_vector_type(16))) _Float16 v16h;
typedef __attribute__((ext_vector_type(8)))  _Float16 v8h;
typedef __attribute__((ext_vector_type(8)))  float    v8f;

constexpr int T = 128, B = 32, V = 10000, H = 256;
constexpr int NTILES = V / 16;          // 625 N-tiles in the dense GEMM
constexpr int LDSW   = H + 8;           // padded row stride (halves) for LDS h buffers

static __device__ __forceinline__ v16h combine16(v8h lo, v8h hi) {
  v16h r;
#pragma unroll
  for (int i = 0; i < 8; ++i) { r[i] = lo[i]; r[i + 8] = hi[i]; }
  return r;
}

// A fragment (16x32 f16, NT layout, K contiguous in memory).
// lane<16: row=lane, K = 0..15 contiguous; lane>=16: same row, K = 16..31.
static __device__ __forceinline__ v16h load_a_frag(const _Float16* rowk, int hi8) {
  const _Float16* p = rowk + hi8 * 2;              // hi8 = 0 or 8
  v8h lo = *(const v8h*)(p);
  v8h hi = *(const v8h*)(p + 8);
  return combine16(lo, hi);
}

// B fragment (32x16 f16). lane<16: col=lane, K=0..15 contiguous; lane>=16: K=16..31.
static __device__ __forceinline__ v16h load_b_frag(const _Float16* rowk, int hi16) {
  const _Float16* p = rowk + hi16;                 // hi16 = 0 or 16
  v8h lo = *(const v8h*)(p);
  v8h hi = *(const v8h*)(p + 8);
  return combine16(lo, hi);
}

// ---------------- kernel 1: embedding gather (+ fold both biases) ----------------
__global__ __launch_bounds__(256) void gather_proj(
    const int* __restrict__ input, const float* __restrict__ w_ih,
    const float* __restrict__ b_ih, const float* __restrict__ b_hh,
    float* __restrict__ proj) {
  int idx = blockIdx.x * 256 + threadIdx.x;        // over T*B*H
  int h  = idx & (H - 1);
  int tb = idx >> 8;
  int tok = input[tb];
  proj[idx] = w_ih[(size_t)h * V + tok] + b_ih[h] + b_hh[h];
}

// ---------------- kernel 2: w_dense f32 -> f16 ----------------
__global__ __launch_bounds__(256) void cvt_wdense(
    const float* __restrict__ wd, _Float16* __restrict__ wd16) {
  int idx = blockIdx.x * 256 + threadIdx.x;        // over V*H
  wd16[idx] = (_Float16)wd[idx];
}

// ---------------- kernel 3: sequential RNN scan, one workgroup, 32 waves ----------------
// Each wave owns one 16x16 tile of the (32 x 256) h_new; w_hh B-fragments are
// time-invariant and live entirely in registers (8 x v16h per wave).
__global__ __launch_bounds__(1024) void rnn_scan(
    const float* __restrict__ proj, const float* __restrict__ h0,
    const float* __restrict__ w_hh, _Float16* __restrict__ ys,
    float* __restrict__ h_last) {
  __shared__ _Float16 sh[2][B * LDSW];             // ping-pong hidden state, ~33 KB

  const int tid = threadIdx.x;
  const int wid = tid >> 5, lane = tid & 31;
  const int lane16 = lane & 15;
  const int hi8    = (lane & 16) ? 8  : 0;
  const int hi16   = (lane & 16) ? 16 : 0;
  const int rowoff = (lane & 16) ? 8  : 0;
  const int mt = wid >> 4;                         // 0..1  (M tile: batch rows)
  const int nt = wid & 15;                         // 0..15 (N tile: hidden cols)
  const int n  = nt * 16 + lane16;                 // this lane's output column

  // Preload all K-chunks of w_hh for this wave's column tile (f32 -> f16).
  v16h whhf[8];
  {
    const float* wr = w_hh + (size_t)n * H;
#pragma unroll
    for (int kc = 0; kc < 8; ++kc) {
      const float* p = wr + kc * 32 + hi16;        // 16 contiguous K values
      v16h f;
#pragma unroll
      for (int i = 0; i < 16; ++i) f[i] = (_Float16)p[i];
      whhf[kc] = f;
    }
  }

  // Load h0 into buffer 0 (f32 -> f16).
  for (int i = tid; i < B * H; i += 1024) {
    int b = i >> 8, c = i & (H - 1);
    sh[0][b * LDSW + c] = (_Float16)h0[i];
  }
  __syncthreads();

  for (int t = 0; t < T; ++t) {
    const int cur = t & 1, nxt = cur ^ 1;
    v8f acc = {};
#pragma unroll
    for (int kc = 0; kc < 8; ++kc) {
      v16h a = load_a_frag(&sh[cur][(mt * 16 + lane16) * LDSW + kc * 32], hi8);
      acc = __builtin_amdgcn_wmma_f32_16x16x32_f16(
          false, a, false, whhf[kc], (short)0, acc, false, false);
    }
    const float* projt = proj + (size_t)t * B * H;
#pragma unroll
    for (int r = 0; r < 8; ++r) {
      int b = mt * 16 + r + rowoff;
      float v = tanhf(acc[r] + projt[b * H + n]);
      _Float16 hv = (_Float16)v;
      sh[nxt][b * LDSW + n] = hv;
      ys[((size_t)t * B + b) * H + n] = hv;
      if (t == T - 1) h_last[b * H + n] = v;
    }
    __syncthreads();
  }
}

// ---------------- kernel 4: dense projection GEMM (4096 x 10000 x 256) ----------------
// One wave computes a 32x64 strip: 2 m-tiles x 4 n-tiles. A-fragments are
// reused across 4 n-tiles; each B-fragment is reused across both m-tiles.
// 64 WMMAs per wave, 6 fragment loads per k-chunk (~22 FLOP/byte from L2).
__global__ __launch_bounds__(256) void dense_gemm(
    const _Float16* __restrict__ ys, const _Float16* __restrict__ wd16,
    const float* __restrict__ b_dense, float* __restrict__ out) {
  const int tid = threadIdx.x, wid = tid >> 5, lane = tid & 31;
  const int lane16 = lane & 15;
  const int hi8    = (lane & 16) ? 8  : 0;
  const int hi16   = (lane & 16) ? 16 : 0;
  const int rowoff = (lane & 16) ? 8  : 0;

  const int task = blockIdx.x * 8 + wid;           // 128 m-groups x 157 n-groups
  const int mg = task & 127;                       // m-group: 2 m-tiles
  const int ng = task >> 7;                        // 0..156: 4 n-tiles

  v8f acc[2][4] = {};
  const _Float16* arow0 = ys + (size_t)(mg * 32 + lane16) * H;       // m-tile 0 row
  const _Float16* arow1 = arow0 + (size_t)16 * H;                    // m-tile 1 row
#pragma unroll
  for (int kc = 0; kc < 8; ++kc) {
    v16h a0 = load_a_frag(arow0 + kc * 32, hi8);
    v16h a1 = load_a_frag(arow1 + kc * 32, hi8);
#pragma unroll
    for (int j = 0; j < 4; ++j) {
      int nt = ng * 4 + j;
      if (nt >= NTILES) continue;                  // uniform per wave
      const _Float16* brow = wd16 + (size_t)(nt * 16 + lane16) * H + kc * 32;
      v16h bf = load_b_frag(brow, hi16);
      acc[0][j] = __builtin_amdgcn_wmma_f32_16x16x32_f16(
          false, a0, false, bf, (short)0, acc[0][j], false, false);
      acc[1][j] = __builtin_amdgcn_wmma_f32_16x16x32_f16(
          false, a1, false, bf, (short)0, acc[1][j], false, false);
    }
  }
#pragma unroll
  for (int j = 0; j < 4; ++j) {
    int nt = ng * 4 + j;
    if (nt >= NTILES) continue;
    int n = nt * 16 + lane16;
    float bias = b_dense[n];
#pragma unroll
    for (int i = 0; i < 2; ++i) {
#pragma unroll
      for (int r = 0; r < 8; ++r) {
        int m = mg * 32 + i * 16 + r + rowoff;
        out[(size_t)m * V + n] = acc[i][j][r] + bias;
      }
    }
  }
}

extern "C" void kernel_launch(void* const* d_in, const int* in_sizes, int n_in,
                              void* d_out, int out_size, void* d_ws, size_t ws_size,
                              hipStream_t stream) {
  const int*   input   = (const int*)  d_in[0];
  const float* h0      = (const float*)d_in[1];
  const float* w_ih    = (const float*)d_in[2];
  const float* w_hh    = (const float*)d_in[3];
  const float* b_ih    = (const float*)d_in[4];
  const float* b_hh    = (const float*)d_in[5];
  const float* w_dense = (const float*)d_in[6];
  const float* b_dense = (const float*)d_in[7];
  float* out = (float*)d_out;

  // Workspace layout
  char* ws = (char*)d_ws;
  float*    proj = (float*)ws;                               // T*B*H f32 = 4 MB
  _Float16* ysh  = (_Float16*)(ws + (size_t)T * B * H * 4);  // T*B*H f16 = 2 MB
  _Float16* wd16 = (_Float16*)(ws + (size_t)T * B * H * 6);  // V*H  f16 = 5 MB

  float* h_last = out + (size_t)T * B * V;                   // second tuple output

  gather_proj<<<(T * B * H) / 256, 256, 0, stream>>>(input, w_ih, b_ih, b_hh, proj);
  cvt_wdense<<<(V * H) / 256, 256, 0, stream>>>(w_dense, wd16);
  rnn_scan<<<1, 1024, 0, stream>>>(proj, h0, w_hh, ysh, h_last);

  constexpr int NGROUPS = (NTILES + 3) / 4;                  // 157
  constexpr int NTASKS  = 128 * NGROUPS;                     // 20096, /8 waves per block
  dense_gemm<<<NTASKS / 8, 256, 0, stream>>>(ysh, wd16, b_dense, out);
}
